// RGCNLayer_83021717832456
// MI455X (gfx1250) — matrix operand order repaired and checked by reading, not compiled, not run on hardware
//
#include <hip/hip_runtime.h>
#include <hip/hip_bf16.h>
#include <stdint.h>

// ---------------------------------------------------------------------------
// RGCN layer for MI455X (gfx1250, wave32, WMMA).
//   msg[e]  = x[src[e]] @ W[rel[e]] + b[rel[e]]        (bf16 WMMA, f32 accum)
//   agg[n]  = max over incoming msgs (0 if no in-edges)
//   out     = x @ W[4] + b[4] + agg
// Edges are counting-sorted by relation so almost every 32-edge wave tile is
// relation-uniform -> fast path with no masking, 2 A-tiles share each B frag.
// ---------------------------------------------------------------------------

typedef __attribute__((ext_vector_type(16))) __bf16 v16bf;
typedef __attribute__((ext_vector_type(8)))  float  v8f;

#define IN_DIM  128
#define OUT_DIM 128
#define NREL    4     // edge relation count
#define RTOT    5     // + self-loop slice
#define WFRAG_ELTS (RTOT * 4 * 8 * 32 * 16)   // r,kchunk,ntile,lane,16 bf16

__device__ __forceinline__ unsigned short f32_to_bf16_rne(float f) {
    union { float f; unsigned u; } c; c.f = f;
    unsigned u = c.u;
    unsigned r = u + 0x7FFFu + ((u >> 16) & 1u);   // round-to-nearest-even
    return (unsigned short)(r >> 16);
}

union AFrag { uint32_t d[8]; uint4 q[2]; v16bf v; };
union BFrag { uint4 q[2]; v16bf v; };

// A fragments, ISA 16-bit A layout: lane's row = lane%16,
// q[0] -> K in [half*8, half*8+7], q[1] -> +16.
__device__ __forceinline__ void load_a(const unsigned short* __restrict__ xb,
                                       long srcOff, int half, AFrag* aF) {
#pragma unroll
    for (int kk = 0; kk < 4; ++kk) {
        const uint4* ap = reinterpret_cast<const uint4*>(xb + srcOff + kk * 32 + half * 8);
        aF[kk].q[0] = ap[0];
        aF[kk].q[1] = ap[2];               // +16 bf16 elements = +2 uint4
    }
}

__device__ __forceinline__ BFrag load_b(const unsigned short* __restrict__ wf,
                                        int r, int kk, int nt, int lane) {
    BFrag b;
    const uint4* bq = reinterpret_cast<const uint4*>(
        wf + ((((r * 4 + kk) * 8 + nt) * 32 + lane) << 4));
    b.q[0] = bq[0]; b.q[1] = bq[1];
    return b;
}

// C layout: VGPR i -> M = i + 8*half, N = lane%16. uniRel>=0: relation uniform.
__device__ __forceinline__ void store_tile_max(v8f c, int myDst, int myRel, int uniRel,
                                               const float* __restrict__ bias,
                                               float* __restrict__ agg,
                                               int nt, int row, int half) {
    int col = nt * 16 + row;
#pragma unroll
    for (int i2 = 0; i2 < 8; ++i2) {
        int rsel = half * 8 + i2;
        int relR = (uniRel >= 0) ? uniRel : __shfl(myRel, rsel, 32);
        int dstR = __shfl(myDst, rsel, 32);
        if (relR < 0) continue;
        float v = c[i2] + bias[relR * OUT_DIM + col];
        atomicMax(agg + (long)dstR * OUT_DIM + col, v);
    }
}

// ---- prep kernels ---------------------------------------------------------

__global__ void k_init(float* __restrict__ agg, int* __restrict__ counts, long total) {
    long i = (long)blockIdx.x * blockDim.x + threadIdx.x;
    if (i < total) agg[i] = -__builtin_inff();
    if (i < 32)    counts[i] = 0;
}

__global__ void k_cvt_x(const float* __restrict__ x, unsigned short* __restrict__ xb, long total) {
    long i = (long)blockIdx.x * blockDim.x + threadIdx.x;
    if (i < total) xb[i] = f32_to_bf16_rne(x[i]);
}

// Pre-swizzle weights into WMMA B-fragment order:
//   elem(r, kk, nt, lane, j) = W[r][kk*32 + (j/8)*16 + (lane/16)*8 + j%8][nt*16 + lane%16]
__global__ void k_swz_w(const float* __restrict__ w, unsigned short* __restrict__ wf, int total) {
    int i = blockIdx.x * blockDim.x + threadIdx.x;
    if (i >= total) return;
    int j    =  i        & 15;
    int lane = (i >> 4)  & 31;
    int nt   = (i >> 9)  & 7;
    int kk   = (i >> 12) & 3;
    int r    =  i >> 14;
    int K    = kk * 32 + (j >> 3) * 16 + (lane >> 4) * 8 + (j & 7);
    int col  = nt * 16 + (lane & 15);
    wf[i] = f32_to_bf16_rne(w[(r * IN_DIM + K) * OUT_DIM + col]);
}

// Counting sort of edge ids by relation -> relation-uniform wave tiles.
__global__ void k_hist(const int* __restrict__ rel, int* __restrict__ counts, int E) {
    int e = blockIdx.x * blockDim.x + threadIdx.x;
    if (e < E) atomicAdd(&counts[rel[e]], 1);
}
__global__ void k_prefix(int* counts) {
    if (threadIdx.x == 0 && blockIdx.x == 0) {
        int s = 0;
        for (int r = 0; r < NREL; ++r) { counts[4 + r] = s; s += counts[r]; }
    }
}
__global__ void k_scatter(const int* __restrict__ rel, int* __restrict__ counts,
                          int* __restrict__ perm, int E) {
    int e = blockIdx.x * blockDim.x + threadIdx.x;
    if (e < E) {
        int p = atomicAdd(&counts[4 + rel[e]], 1);
        perm[p] = e;
    }
}

// ---- per-edge message WMMA + atomic-max scatter ---------------------------
// One wave handles 32 edges (two 16-row A tiles sharing every B fragment).

__global__ void __launch_bounds__(256)
k_msg(const unsigned short* __restrict__ xb,
      const unsigned short* __restrict__ wf,
      const float* __restrict__ bias,
      const int* __restrict__ src, const int* __restrict__ dst,
      const int* __restrict__ rel, const int* __restrict__ perm,
      float* __restrict__ agg, int nwaves, int E)
{
    int wave = (int)((blockIdx.x * blockDim.x + threadIdx.x) >> 5);
    int lane = threadIdx.x & 31;
    if (wave >= nwaves) return;            // wave-uniform exit (EXEC all-1 inside)
    int row  = lane & 15;
    int half = lane >> 4;
    int base = wave * 32;

    // two edge rows per lane: tile0 = rows base..base+15, tile1 = +16
    int idx0 = base + row,            idx1 = base + 16 + row;
    int ed0  = perm[idx0 < E ? idx0 : E - 1];
    int ed1  = perm[idx1 < E ? idx1 : E - 1];
    int myRel0 = (idx0 < E) ? rel[ed0] : -1;
    int myRel1 = (idx1 < E) ? rel[ed1] : -1;
    int myDst0 = dst[ed0], myDst1 = dst[ed1];
    long srcOff0 = (long)src[ed0] * IN_DIM;
    long srcOff1 = (long)src[ed1] * IN_DIM;

    AFrag aF0[4], aF1[4];
    load_a(xb, srcOff0, half, aF0);
    load_a(xb, srcOff1, half, aF1);

    int relF = __shfl(myRel0, 0, 32);
    bool uniform = relF >= 0 &&
                   (__ballot(myRel0 != relF) | __ballot(myRel1 != relF)) == 0;

    if (uniform) {
        // ---- fast path: no masking, B fragment reused by both A tiles ----
        for (int nt = 0; nt < 8; ++nt) {
            BFrag bFs[4];
#pragma unroll
            for (int kk = 0; kk < 4; ++kk) bFs[kk] = load_b(wf, relF, kk, nt, lane);
            v8f c0 = {}, c1 = {};
#pragma unroll
            for (int kk = 0; kk < 4; ++kk) {
                c0 = __builtin_amdgcn_wmma_f32_16x16x32_bf16(
                         false, aF0[kk].v, false, bFs[kk].v, (short)0, c0, false, false);
                c1 = __builtin_amdgcn_wmma_f32_16x16x32_bf16(
                         false, aF1[kk].v, false, bFs[kk].v, (short)0, c1, false, false);
            }
            store_tile_max(c0, myDst0, myRel0, relF, bias, agg, nt, row, half);
            store_tile_max(c1, myDst1, myRel1, relF, bias, agg, nt, row, half);
        }
    } else {
        // ---- cold path (bin-boundary waves): per-relation masked WMMA ----
#pragma unroll 1
        for (int t = 0; t < 2; ++t) {
            const AFrag* aF = t ? aF1 : aF0;
            int myRel = t ? myRel1 : myRel0;
            int myDst = t ? myDst1 : myDst0;
#pragma unroll 1
            for (int nt = 0; nt < 8; ++nt) {
                v8f c = {};
#pragma unroll 1
                for (int r = 0; r < NREL; ++r) {
                    if (__ballot(myRel == r) == 0) continue;
                    uint32_t msk = (myRel == r) ? 0xFFFFFFFFu : 0u;
#pragma unroll
                    for (int kk = 0; kk < 4; ++kk) {
                        BFrag bF = load_b(wf, r, kk, nt, lane);
                        AFrag am;
#pragma unroll
                        for (int d2 = 0; d2 < 8; ++d2) am.d[d2] = aF[kk].d[d2] & msk;
                        c = __builtin_amdgcn_wmma_f32_16x16x32_bf16(
                                false, am.v, false, bF.v, (short)0, c, false, false);
                    }
                }
                store_tile_max(c, myDst, myRel, -1, bias, agg, nt, row, half);
            }
        }
    }
}

// ---- self-loop GEMM + combine ---------------------------------------------

__global__ void __launch_bounds__(256)
k_self(const unsigned short* __restrict__ xb,
       const unsigned short* __restrict__ wf,
       const float* __restrict__ bias,
       const float* __restrict__ agg,
       float* __restrict__ out, int ntiles, long N)
{
    int wave = (int)((blockIdx.x * blockDim.x + threadIdx.x) >> 5);
    int lane = threadIdx.x & 31;
    if (wave >= ntiles) return;
    int row  = lane & 15;
    int half = lane >> 4;

    long node  = (long)wave * 16 + row;
    long cnode = node < N ? node : N - 1;

    AFrag aF[4];
    load_a(xb, cnode * IN_DIM, half, aF);

    for (int nt = 0; nt < 8; ++nt) {
        BFrag bFs[4];
#pragma unroll
        for (int kk = 0; kk < 4; ++kk) bFs[kk] = load_b(wf, NREL, kk, nt, lane);
        v8f c = {};
#pragma unroll
        for (int kk = 0; kk < 4; ++kk)
            c = __builtin_amdgcn_wmma_f32_16x16x32_bf16(
                    false, aF[kk].v, false, bFs[kk].v, (short)0, c, false, false);
        int col = nt * 16 + row;
#pragma unroll
        for (int i2 = 0; i2 < 8; ++i2) {
            long nrow = (long)wave * 16 + half * 8 + i2;
            if (nrow >= N) continue;
            float a = agg[nrow * OUT_DIM + col];
            if (a < -1e37f) a = 0.0f;      // nodes with no in-edges -> 0
            out[nrow * OUT_DIM + col] = c[i2] + bias[NREL * OUT_DIM + col] + a;
        }
    }
}

// ---- host launcher --------------------------------------------------------

extern "C" void kernel_launch(void* const* d_in, const int* in_sizes, int n_in,
                              void* d_out, int out_size, void* d_ws, size_t ws_size,
                              hipStream_t stream) {
    const float* x    = (const float*)d_in[0];
    const float* w    = (const float*)d_in[1];
    const float* bias = (const float*)d_in[2];
    const int*   src  = (const int*)d_in[3];
    const int*   dst  = (const int*)d_in[4];
    const int*   rel  = (const int*)d_in[5];

    long N = in_sizes[0] / IN_DIM;
    int  E = in_sizes[3];

    // workspace layout
    char* ws = (char*)d_ws;
    size_t off = 0;
    float* agg = (float*)(ws + off);            off += (size_t)N * OUT_DIM * sizeof(float);
    unsigned short* xb = (unsigned short*)(ws + off); off += (size_t)N * IN_DIM * 2;
    off = (off + 255) & ~(size_t)255;
    unsigned short* wfr = (unsigned short*)(ws + off); off += (size_t)WFRAG_ELTS * 2;
    off = (off + 255) & ~(size_t)255;
    int* counts = (int*)(ws + off);             off += 128;
    int* perm = (int*)(ws + off);               off += (size_t)E * sizeof(int);

    const int thr = 256;
    long totNF = N * (long)OUT_DIM;
    long totNI = N * (long)IN_DIM;

    k_init   <<<(unsigned)((totNF + thr - 1) / thr), thr, 0, stream>>>(agg, counts, totNF);
    k_cvt_x  <<<(unsigned)((totNI + thr - 1) / thr), thr, 0, stream>>>(x, xb, totNI);
    k_swz_w  <<<(WFRAG_ELTS + thr - 1) / thr, thr, 0, stream>>>(w, wfr, WFRAG_ELTS);
    k_hist   <<<(E + thr - 1) / thr, thr, 0, stream>>>(rel, counts, E);
    k_prefix <<<1, 1, 0, stream>>>(counts);
    k_scatter<<<(E + thr - 1) / thr, thr, 0, stream>>>(rel, counts, perm, E);

    int nwaves = (E + 31) / 32;                 // 32 edges per wave
    k_msg<<<(nwaves + 7) / 8, 256, 0, stream>>>(xb, wfr, bias, src, dst, rel, perm,
                                                agg, nwaves, E);

    int ntiles = (int)((N + 15) / 16);          // 16 nodes per wave
    k_self<<<(ntiles + 7) / 8, 256, 0, stream>>>(xb, wfr, bias, agg,
                                                 (float*)d_out, ntiles, N);
}
